// MNISTBNN_72335839199898
// MI455X (gfx1250) — 8-bit matrix operand order repaired and checked by reading, compiled, not run on hardware
//
#include <hip/hip_runtime.h>
#include <cstdint>
#include <cstddef>

typedef __attribute__((ext_vector_type(8))) int   v8i;
typedef __attribute__((ext_vector_type(4))) int   v4i;

#define BATCH 2048
#define BN_EPS 1e-5f

#define H1P 30
#define H2P 18

static const size_t OFF_ACT1 = 0;
static const size_t OFF_ACT2 = OFF_ACT1 + (size_t)BATCH*H1P*H1P*32;
static const size_t OFF_ACT3 = OFF_ACT2 + (size_t)BATCH*H2P*H2P*64;
static const size_t OFF_WF2  = OFF_ACT3 + (size_t)BATCH*3136;
static const size_t OFF_WF3  = OFF_WF2  + 4*5*32*8*4;
static const size_t OFF_WFC  = OFF_WF3  + 4*9*32*8*4;
static const size_t OFF_BN   = OFF_WFC  + 16*49*32*8*4;
static const size_t OFF_FCO  = OFF_BN   + 1024*4;


__global__ void k_zero(uint8_t* __restrict__ ws, size_t n16) {
    size_t i = (size_t)blockIdx.x * blockDim.x + threadIdx.x;
    if (i < n16) {
        v4i z = {0, 0, 0, 0};
        ((v4i*)ws)[i] = z;
    }
}

__global__ void k_bnprep(const float* g1, const float* b1, const float* m1, const float* v1,
                         const float* g2, const float* b2, const float* m2, const float* v2,
                         const float* g3, const float* b3, const float* m3, const float* v3,
                         const float* g4, const float* b4, const float* m4, const float* v4,
                         float* __restrict__ bn) {
    int i = threadIdx.x;
    if (i < 32) {
        float s = g1[i] * rsqrtf(v1[i] + BN_EPS);
        bn[i] = s; bn[32 + i] = b1[i] - m1[i] * s;
    } else if (i < 96) {
        int j = i - 32;
        float s = g2[j] * rsqrtf(v2[j] + BN_EPS);
        bn[64 + j] = s; bn[128 + j] = b2[j] - m2[j] * s;
    } else if (i < 160) {
        int j = i - 96;
        float s = g3[j] * rsqrtf(v3[j] + BN_EPS);
        bn[192 + j] = s; bn[256 + j] = b3[j] - m3[j] * s;
    } else if (i < 416) {
        int j = i - 160;
        float s = g4[j] * rsqrtf(v4[j] + BN_EPS);
        bn[320 + j] = s; bn[576 + j] = b4[j] - m4[j] * s;
    }
}

__device__ __forceinline__ uint32_t pack_sign4(float a, float b, float c, float d) {
    uint32_t r;
    r  = (uint32_t)(uint8_t)((a >= 0.f) ? 1 : -1);
    r |= (uint32_t)(uint8_t)((b >= 0.f) ? 1 : -1) << 8;
    r |= (uint32_t)(uint8_t)((c >= 0.f) ? 1 : -1) << 16;
    r |= (uint32_t)(uint8_t)((d >= 0.f) ? 1 : -1) << 24;
    return r;
}

__global__ void k_wprep(const float* __restrict__ w2, const float* __restrict__ w3,
                        const float* __restrict__ wfc,
                        int* __restrict__ wf2, int* __restrict__ wf3, int* __restrict__ wffc) {
    const int NA = 4 * 5 * 32 * 8;
    const int NB = 4 * 9 * 32 * 8;
    const int NC = 16 * 49 * 32 * 8;
    int idx = blockIdx.x * blockDim.x + threadIdx.x;
    if (idx >= NA + NB + NC) return;

    if (idx < NA) {
        int r = idx;
        int v = r & 7, lane = (r >> 3) & 31, ks = (r >> 8) % 5, nt = r / 1280;
        int h = lane >> 4, n = nt * 16 + (lane & 15);
        int kb = ks * 64 + (v >> 1) * 16 + h * 8 + (v & 1) * 4;
        uint32_t out = 0;
        for (int j = 0; j < 4; j++) {
            int k = kb + j;
            int8_t s = 0;
            if (k < 288) {
                int p = k >> 5, c = k & 31, dy = p / 3, dx = p % 3;
                float wv = w2[((n * 32 + c) * 3 + dy) * 3 + dx];
                s = (wv >= 0.f) ? 1 : -1;
            }
            out |= (uint32_t)(uint8_t)s << (8 * j);
        }
        wf2[idx] = (int)out;
    } else if (idx < NA + NB) {
        int r = idx - NA;
        int v = r & 7, lane = (r >> 3) & 31, ks = (r >> 8) % 9, nt = r / 2304;
        int h = lane >> 4, n = nt * 16 + (lane & 15);
        int kb = ks * 64 + (v >> 1) * 16 + h * 8 + (v & 1) * 4;
        uint32_t out = 0;
        for (int j = 0; j < 4; j++) {
            int k = kb + j;
            int p = k >> 6, c = k & 63, dy = p / 3, dx = p % 3;
            float wv = w3[((n * 64 + c) * 3 + dy) * 3 + dx];
            out |= (uint32_t)(uint8_t)(int8_t)((wv >= 0.f) ? 1 : -1) << (8 * j);
        }
        wf3[r] = (int)out;
    } else {
        int r = idx - NA - NB;
        int v = r & 7, lane = (r >> 3) & 31, ks = (r >> 8) % 49, nt = r / 12544;
        int h = lane >> 4, n = nt * 16 + (lane & 15);
        int kb = ks * 64 + (v >> 1) * 16 + h * 8 + (v & 1) * 4;
        const float* wr = wfc + (size_t)n * 3136 + kb;
        wffc[r] = (int)pack_sign4(wr[0], wr[1], wr[2], wr[3]);
    }
}

__global__ void __launch_bounds__(128) k_stem(const float* __restrict__ x,
                                              const float* __restrict__ wst,
                                              const float* __restrict__ bn,
                                              int8_t* __restrict__ act1) {
    __shared__ float sw[288];
    __shared__ float ss[32], sb[32];
    int tid = threadIdx.x;
    for (int i = tid; i < 288; i += 128) sw[i] = wst[i];
    if (tid < 32) { ss[tid] = bn[tid]; sb[tid] = bn[32 + tid]; }
    __syncthreads();

    int gid = blockIdx.x * 128 + tid;
    int px = gid % 28, py = (gid / 28) % 28, b = gid / 784;
    const float* xb = x + (size_t)b * 784;

    float win[9];
#pragma unroll
    for (int dy = 0; dy < 3; dy++)
#pragma unroll
        for (int dx = 0; dx < 3; dx++) {
            int yy = py + dy - 1, xx = px + dx - 1;
            win[dy * 3 + dx] =
                (yy >= 0 && yy < 28 && xx >= 0 && xx < 28) ? xb[yy * 28 + xx] : 0.f;
        }

    uint32_t out[8];
#pragma unroll
    for (int co = 0; co < 32; co++) {
        float acc = 0.f;
#pragma unroll
        for (int t = 0; t < 9; t++) acc = fmaf(sw[co * 9 + t], win[t], acc);
        float yb = fmaf(acc, ss[co], sb[co]);
        int8_t s = (yb >= 0.f) ? 1 : -1;
        if ((co & 3) == 0) out[co >> 2] = 0;
        out[co >> 2] |= (uint32_t)(uint8_t)s << (8 * (co & 3));
    }
    v4i* dst = (v4i*)(act1 + ((size_t)(b * H1P + py + 1) * H1P + (px + 1)) * 32);
    dst[0] = ((v4i*)out)[0];
    dst[1] = ((v4i*)out)[1];
}

__global__ void __launch_bounds__(128) k_conv2(const int8_t* __restrict__ act1,
                                               const int* __restrict__ wf2,
                                               const float* __restrict__ bn,
                                               int8_t* __restrict__ act2) {
    __shared__ __align__(16) int8_t at[16 * 320];
    int tid = threadIdx.x;
    int bidx = blockIdx.x;
    int tx = bidx % 7, ty = (bidx / 7) % 7, b = bidx / 49;
    int Y0 = 4 * ty, X0 = 4 * tx;

    for (int i = tid; i < 288; i += 128) {
        int chunk = i % 6, seg = i / 6, dy = seg % 3, m = seg / 3;
        int py = (m >> 2) + Y0, px = (m & 3) + X0;
        const v4i* src = (const v4i*)(act1 +
            ((size_t)(b * H1P + py + dy) * H1P + px) * 32 + chunk * 16);
        *(v4i*)(at + m * 320 + dy * 96 + chunk * 16) = *src;
    }
    if (tid < 32) {
        v4i z = {0, 0, 0, 0};
        *(v4i*)(at + (tid >> 1) * 320 + 288 + (tid & 1) * 16) = z;
    }
    __syncthreads();

    int w = tid >> 5, lane = tid & 31;
    int h = lane >> 4, n16 = lane & 15;
    v8i c = {0, 0, 0, 0, 0, 0, 0, 0};
#pragma unroll
    for (int ks = 0; ks < 5; ks++) {
        v8i bfr = *(const v8i*)(wf2 + (size_t)((w * 5 + ks) * 32 + lane) * 8);
        v8i a;
#pragma unroll
        for (int v = 0; v < 8; v++) {
            int kb = ks * 64 + (v >> 1) * 16 + h * 8 + (v & 1) * 4;
            a[v] = *(const int*)(at + (lane & 15) * 320 + kb);
        }
        c = __builtin_amdgcn_wmma_i32_16x16x64_iu8(true, a, true, bfr, c, false, false);
    }

    int ch = w * 16 + n16;
    float sc = bn[64 + ch], bi = bn[128 + ch];
#pragma unroll
    for (int qx = 0; qx < 2; qx++) {
        int v0 = 2 * qx;
        float p0 = fmaf((float)c[v0],     sc, bi);
        float p1 = fmaf((float)c[v0 + 1], sc, bi);
        float p2 = fmaf((float)c[v0 + 4], sc, bi);
        float p3 = fmaf((float)c[v0 + 5], sc, bi);
        float mx = fmaxf(fmaxf(p0, p1), fmaxf(p2, p3));
        int8_t s = (mx >= 0.f) ? 1 : -1;
        int oy = (Y0 >> 1) + h, ox = (X0 >> 1) + qx;
        act2[((size_t)(b * H2P + oy + 1) * H2P + ox + 1) * 64 + ch] = s;
    }
}

__global__ void __launch_bounds__(128) k_conv3(const int8_t* __restrict__ act2,
                                               const int* __restrict__ wf3,
                                               const float* __restrict__ bn,
                                               int8_t* __restrict__ act3) {
    __shared__ __align__(16) int8_t at[16 * 576];
    int tid = threadIdx.x;
    int bidx = blockIdx.x;
    int tx = bidx % 4, ty = (bidx / 4) % 4, b = bidx / 16;
    int Y0 = 4 * ty, X0 = 4 * tx;

    for (int i = tid; i < 576; i += 128) {
        int chunk = i % 12, seg = i / 12, dy = seg % 3, m = seg / 3;
        int py = (m >> 2) + Y0, px = (m & 3) + X0;
        const v4i* src = (const v4i*)(act2 +
            ((size_t)(b * H2P + py + dy) * H2P + px) * 64 + chunk * 16);
        *(v4i*)(at + m * 576 + dy * 192 + chunk * 16) = *src;
    }
    __syncthreads();

    int w = tid >> 5, lane = tid & 31;
    int h = lane >> 4, n16 = lane & 15;
    v8i c = {0, 0, 0, 0, 0, 0, 0, 0};
#pragma unroll
    for (int ks = 0; ks < 9; ks++) {
        v8i bfr = *(const v8i*)(wf3 + (size_t)((w * 9 + ks) * 32 + lane) * 8);
        v8i a;
#pragma unroll
        for (int v = 0; v < 8; v++) {
            int kb = ks * 64 + (v >> 1) * 16 + h * 8 + (v & 1) * 4;
            a[v] = *(const int*)(at + (lane & 15) * 576 + kb);
        }
        c = __builtin_amdgcn_wmma_i32_16x16x64_iu8(true, a, true, bfr, c, false, false);
    }

    int ch = w * 16 + n16;
    float sc = bn[192 + ch], bi = bn[256 + ch];
#pragma unroll
    for (int qx = 0; qx < 2; qx++) {
        int v0 = 2 * qx;
        float p0 = fmaf((float)c[v0],     sc, bi);
        float p1 = fmaf((float)c[v0 + 1], sc, bi);
        float p2 = fmaf((float)c[v0 + 4], sc, bi);
        float p3 = fmaf((float)c[v0 + 5], sc, bi);
        float mx = fmaxf(fmaxf(p0, p1), fmaxf(p2, p3));
        int8_t s = (mx >= 0.f) ? 1 : -1;
        int oy = (Y0 >> 1) + h, ox = (X0 >> 1) + qx;
        if (oy < 7 && ox < 7)
            act3[((size_t)(b * 64 + ch) * 7 + oy) * 7 + ox] = s;
    }
}

__global__ void __launch_bounds__(128) k_fc(const int8_t* __restrict__ act3,
                                            const int* __restrict__ wffc,
                                            const float* __restrict__ bn,
                                            float* __restrict__ fco) {
    int tid = threadIdx.x;
    int w = tid >> 5, lane = tid & 31;
    int h = lane >> 4, n16 = lane & 15;
    int nt = blockIdx.y * 4 + w;
    int M0 = blockIdx.x * 16;
    const int8_t* arow = act3 + (size_t)(M0 + (lane & 15)) * 3136;

    v8i c = {0, 0, 0, 0, 0, 0, 0, 0};
    for (int ks = 0; ks < 49; ks++) {
        v8i bfr = *(const v8i*)(wffc + (size_t)((nt * 49 + ks) * 32 + lane) * 8);
        v8i a;
#pragma unroll
        for (int v = 0; v < 8; v++) {
            int kb = ks * 64 + (v >> 1) * 16 + h * 8 + (v & 1) * 4;
            a[v] = *(const int*)(arow + kb);
        }
        c = __builtin_amdgcn_wmma_i32_16x16x64_iu8(true, a, true, bfr, c, false, false);
    }

    int col = nt * 16 + n16;
    float sc = bn[320 + col], bi = bn[576 + col];
#pragma unroll
    for (int v = 0; v < 8; v++) {
        int row = M0 + v + h * 8;
        fco[(size_t)row * 256 + col] = fmaf((float)c[v], sc, bi);
    }
}

__global__ void k_head(const float* __restrict__ fco, const float* __restrict__ wh,
                       const float* __restrict__ bh, float* __restrict__ out) {
    int idx = blockIdx.x * blockDim.x + threadIdx.x;
    if (idx >= BATCH * 10) return;
    int o = idx % 10, b = idx / 10;
    const float* r = fco + (size_t)b * 256;
    const float* ww = wh + (size_t)o * 256;
    float s = bh[o];
#pragma unroll 4
    for (int k = 0; k < 256; k++) s = fmaf(r[k], ww[k], s);
    out[idx] = s;
}

extern "C" void kernel_launch(void* const* d_in, const int* in_sizes, int n_in,
                              void* d_out, int out_size, void* d_ws, size_t ws_size,
                              hipStream_t stream) {
    const float* x      = (const float*)d_in[0];
    const float* w_stem = (const float*)d_in[1];
    const float* g1 = (const float*)d_in[2],  *b1 = (const float*)d_in[3];
    const float* m1 = (const float*)d_in[4],  *v1 = (const float*)d_in[5];
    const float* w2 = (const float*)d_in[6];
    const float* g2 = (const float*)d_in[7],  *b2 = (const float*)d_in[8];
    const float* m2 = (const float*)d_in[9],  *v2 = (const float*)d_in[10];
    const float* w3 = (const float*)d_in[11];
    const float* g3 = (const float*)d_in[12], *b3 = (const float*)d_in[13];
    const float* m3 = (const float*)d_in[14], *v3 = (const float*)d_in[15];
    const float* w_fc = (const float*)d_in[16];
    const float* g4 = (const float*)d_in[17], *b4 = (const float*)d_in[18];
    const float* m4 = (const float*)d_in[19], *v4 = (const float*)d_in[20];
    const float* w_head = (const float*)d_in[21];
    const float* b_head = (const float*)d_in[22];

    uint8_t* ws = (uint8_t*)d_ws;
    int8_t* act1 = (int8_t*)(ws + OFF_ACT1);
    int8_t* act2 = (int8_t*)(ws + OFF_ACT2);
    int8_t* act3 = (int8_t*)(ws + OFF_ACT3);
    int*    wf2  = (int*)(ws + OFF_WF2);
    int*    wf3  = (int*)(ws + OFF_WF3);
    int*    wffc = (int*)(ws + OFF_WFC);
    float*  bn   = (float*)(ws + OFF_BN);
    float*  fco  = (float*)(ws + OFF_FCO);

    size_t zbytes = (size_t)BATCH * H1P * H1P * 32 + (size_t)BATCH * H2P * H2P * 64;
    size_t n16 = zbytes / 16;
    k_zero<<<(unsigned)((n16 + 255) / 256), 256, 0, stream>>>(ws, n16);

    k_bnprep<<<1, 416, 0, stream>>>(g1, b1, m1, v1, g2, b2, m2, v2,
                                    g3, b3, m3, v3, g4, b4, m4, v4, bn);

    const int NTOT = 4 * 5 * 32 * 8 + 4 * 9 * 32 * 8 + 16 * 49 * 32 * 8;
    k_wprep<<<(NTOT + 255) / 256, 256, 0, stream>>>(w2, w3, w_fc, wf2, wf3, wffc);

    k_stem<<<BATCH * 784 / 128, 128, 0, stream>>>(x, w_stem, bn, act1);

    k_conv2<<<BATCH * 49, 128, 0, stream>>>(act1, wf2, bn, act2);

    k_conv3<<<BATCH * 16, 128, 0, stream>>>(act2, wf3, bn, act3);

    k_fc<<<dim3(BATCH / 16, 4), 128, 0, stream>>>(act3, wffc, bn, fco);

    k_head<<<(BATCH * 10 + 255) / 256, 256, 0, stream>>>(fco, w_head, b_head, (float*)d_out);
}
